// MultiHeadAttention_73254962200803
// MI455X (gfx1250) — compile-verified, matrix operand order checked by reading, not run on hardware
//
#include <hip/hip_runtime.h>

#define F     1024   // feature dim
#define NH    16     // heads
#define DHEAD 64     // d_k = d_v
#define BATCH 8
#define SEQ   1024

typedef __attribute__((ext_vector_type(16))) __bf16 v16bf;
typedef __attribute__((ext_vector_type(8)))  float  v8f;
typedef __attribute__((ext_vector_type(4)))  unsigned int tdm_v4u;
typedef __attribute__((ext_vector_type(8)))  int          tdm_v8i;
typedef __attribute__((ext_vector_type(4)))  int          tdm_v4i;

// Native fptrunc: lowers to hardware bf16 convert on gfx1250 (pairable into
// packed converts), replacing the 4-op bit-twiddle sequence.
__device__ __forceinline__ __bf16 f2bf(float f) { return (__bf16)f; }

// K-offset of bf16 element pair (2j, 2j+1) for A/B fragments of
// v_wmma_f32_16x16x32_bf16, per CDNA5 ISA 16-bit A-matrix layout.
__device__ __forceinline__ int kmap(int j, int half) {
  return ((j < 4) ? 2 * j : 16 + 2 * (j - 4)) + 8 * half;
}

__device__ __forceinline__ v8f wmma_bf16(v16bf a, v16bf b, v8f c) {
  return __builtin_amdgcn_wmma_f32_16x16x32_bf16(false, a, false, b, (short)0, c,
                                                 false, false);
}

__device__ __forceinline__ unsigned lds_off(const void* p) {
  // generic LDS pointer: low 32 bits = byte offset within workgroup LDS
  return (unsigned)(uintptr_t)p;
}

// ---------------------------------------------------------------------------
// Tensor Data Mover: 2D tile (tile_d0 elems/row x tile_d1 rows) from a
// row-major tensor (row stride stride_d0 elems) into LDS, packed contiguous.
// D# encoding per CDNA5 ISA (group0/group1 bitfields).
// Issue from ONE wave only; completion via s_wait_tensorcnt.
// ---------------------------------------------------------------------------
__device__ __forceinline__ void tdm_load_2d(unsigned lds_addr, const void* gaddr,
                                            unsigned data_size_log2,  // 0=1B,1=2B,2=4B
                                            unsigned tile_d0, unsigned tile_d1,
                                            unsigned tensor_d0, unsigned tensor_d1,
                                            unsigned stride_d0) {
  unsigned long long ga = (unsigned long long)(uintptr_t)gaddr;
  tdm_v4u g0;
  g0[0] = 1u;                                              // count=1, no gather
  g0[1] = lds_addr;                                        // lds_addr[31:0]
  g0[2] = (unsigned)(ga & 0xFFFFFFFFu);                    // global_addr[31:0]
  g0[3] = (unsigned)((ga >> 32) & 0x1FFFFFFu)              // global_addr[56:32]
          | 0x80000000u;                                   // type=2 (image)
  tdm_v8i g1;
  g1[0] = (int)(data_size_log2 << 16);                     // wg_mask=0, data_size
  g1[1] = (int)((tensor_d0 & 0xFFFFu) << 16);              // tensor_dim0[15:0]
  g1[2] = (int)(((tensor_d0 >> 16) & 0xFFFFu) |            // tensor_dim0[31:16]
                ((tensor_d1 & 0xFFFFu) << 16));            // tensor_dim1[15:0]
  g1[3] = (int)(((tensor_d1 >> 16) & 0xFFFFu) |            // tensor_dim1[31:16]
                ((tile_d0 & 0xFFFFu) << 16));              // tile_dim0
  g1[4] = (int)(tile_d1 & 0xFFFFu);                        // tile_dim1 (tile_dim2=0)
  g1[5] = (int)stride_d0;                                  // tensor_dim0_stride[31:0]
  g1[6] = 0;                                               // stride hi / dim1_stride lo
  g1[7] = 0;                                               // dim1_stride hi
  tdm_v4i z = {0, 0, 0, 0};
#if __has_include(<hip/amd_detail/amd_gfx1250_TDM.h>)
  tdm_v8i z8 = {0, 0, 0, 0, 0, 0, 0, 0};
  __builtin_amdgcn_tensor_load_to_lds(g0, g1, z, z, z8, 0);   // therock 6-arg
#else
  __builtin_amdgcn_tensor_load_to_lds(g0, g1, z, z, 0);       // ROCm 7.2 5-arg
#endif
}

// ---------------------------------------------------------------------------
// Kernel 1: QKV projections.  X[8192,1024] @ W[1024,1024] -> bf16 [B,H,S,64].
// grid = (N/64, M/128, 3), block = 256 (8 waves). Q is pre-scaled by 1/8.
// Double-buffered TDM staging: wave 0 issues tile it+1 while the block
// computes tile it (s_wait_tensorcnt 2 keeps the next pair in flight).
// ---------------------------------------------------------------------------
__global__ __launch_bounds__(256) void qkv_proj_kernel(
    const float* __restrict__ Xq, const float* __restrict__ Xk,
    const float* __restrict__ Xv, const float* __restrict__ Wq,
    const float* __restrict__ Wk, const float* __restrict__ Wv,
    __bf16* __restrict__ Qb, __bf16* __restrict__ Kb, __bf16* __restrict__ Vb) {
  __shared__ __align__(16) float AlF[2][128 * 32];   // 2 x 16KB
  __shared__ __align__(16) float BlF[2][32 * 64];    // 2 x 8KB

  const int z = blockIdx.z;
  const float* X = (z == 0) ? Xq : (z == 1) ? Xk : Xv;
  const float* W = (z == 0) ? Wq : (z == 1) ? Wk : Wv;
  __bf16* Dst    = (z == 0) ? Qb : (z == 1) ? Kb : Vb;
  const float oscale = (z == 0) ? 0.125f : 1.0f;   // 1/sqrt(64) folded into Q

  const int tid = threadIdx.x;
  const int w = tid >> 5, lane = tid & 31;
  const int half = lane >> 4, ln = lane & 15;
  const int m0 = blockIdx.y * 128;
  const int n0 = blockIdx.x * 64;

  v8f acc[4];
#pragma unroll
  for (int nt = 0; nt < 4; ++nt) acc[nt] = (v8f){0.f, 0.f, 0.f, 0.f, 0.f, 0.f, 0.f, 0.f};

  if (tid < 32) {   // prologue: stage tile 0
    tdm_load_2d(lds_off(AlF[0]), X + (size_t)m0 * F, 2, 32, 128, F, BATCH * SEQ, F);
    tdm_load_2d(lds_off(BlF[0]), W + n0, 2, 64, 32, F, F, F);
  }

  for (int it = 0; it < F / 32; ++it) {
    const int buf = it & 1;
    __syncthreads();                 // everyone done reading buf^1 (iter it-1)
    if (tid < 32) {
      if (it + 1 < F / 32) {
        const int k1 = (it + 1) * 32;
        tdm_load_2d(lds_off(AlF[buf ^ 1]), X + (size_t)m0 * F + k1, 2,
                    32, 128, F, BATCH * SEQ, F);
        tdm_load_2d(lds_off(BlF[buf ^ 1]), W + (size_t)k1 * F + n0, 2,
                    64, 32, F, F, F);
        __builtin_amdgcn_s_wait_tensorcnt(2);   // tile 'it' landed; it+1 in flight
      } else {
        __builtin_amdgcn_s_wait_tensorcnt(0);
      }
    }
    __syncthreads();

    const float* Af = AlF[buf];
    const float* Bf = BlF[buf];
    v16bf a;
#pragma unroll
    for (int j = 0; j < 8; ++j) {
      int kb = kmap(j, half);
      a[2 * j]     = f2bf(Af[(w * 16 + ln) * 32 + kb]);
      a[2 * j + 1] = f2bf(Af[(w * 16 + ln) * 32 + kb + 1]);
    }
#pragma unroll
    for (int nt = 0; nt < 4; ++nt) {
      v16bf bb;
#pragma unroll
      for (int j = 0; j < 8; ++j) {
        int kb = kmap(j, half);
        bb[2 * j]     = f2bf(Bf[kb * 64 + nt * 16 + ln]);
        bb[2 * j + 1] = f2bf(Bf[(kb + 1) * 64 + nt * 16 + ln]);
      }
      acc[nt] = wmma_bf16(a, bb, acc[nt]);
    }
  }

  // scatter C into [B,H,S,64] bf16
#pragma unroll
  for (int nt = 0; nt < 4; ++nt) {
#pragma unroll
    for (int g = 0; g < 8; ++g) {
      int m = m0 + w * 16 + g + 8 * half;
      int n = n0 + nt * 16 + ln;
      int b = m >> 10, s = m & 1023;
      int h = n >> 6,  d = n & 63;
      Dst[(((size_t)(b * NH + h) * SEQ + s) * DHEAD) + d] = f2bf(acc[nt][g] * oscale);
    }
  }
}

// ---------------------------------------------------------------------------
// Kernel 2: attention. One wave = one (b,h,q-tile of 16). Two-pass online
// softmax; writes attn (fp32, the mandatory 512MB) once; fuses attn@V via a
// per-wave LDS transpose of P into WMMA A-layout.
// grid = B*H*(S/16)/8 = 1024 blocks, block = 256.
// ---------------------------------------------------------------------------
__global__ __launch_bounds__(256) void attn_kernel(
    const __bf16* __restrict__ Qb, const __bf16* __restrict__ Kb,
    const __bf16* __restrict__ Vb, const float* __restrict__ mask,
    float* __restrict__ attn_out, __bf16* __restrict__ Ctx) {
  __shared__ __bf16 Pl[8][16][34];   // per-wave P tile (16 q x 32 k)

  const int tid = threadIdx.x;
  const int w = tid >> 5, lane = tid & 31;
  const int half = lane >> 4, ln = lane & 15;
  const int gw = blockIdx.x * 8 + w;
  const int bh = gw >> 6;           // /(S/16)
  const int qt = gw & 63;
  const int b  = bh >> 4;
  const int h  = bh & 15;
  const int q0 = qt * 16;

  const __bf16* Qp = Qb + (size_t)bh * SEQ * DHEAD;
  const __bf16* Kp = Kb + (size_t)bh * SEQ * DHEAD;
  const __bf16* Vp = Vb + (size_t)bh * SEQ * DHEAD;
  const float*  Mp = mask + (size_t)b * SEQ * SEQ;
  float*        Ap = attn_out + (size_t)bh * SEQ * SEQ;

  // Q fragments (held in registers across both passes)
  v16bf qa0, qa1;
  {
    const __bf16* qr = Qp + (size_t)(q0 + ln) * DHEAD;
#pragma unroll
    for (int j = 0; j < 8; ++j) {
      int kb = kmap(j, half);
      qa0[2 * j]     = qr[kb];
      qa0[2 * j + 1] = qr[kb + 1];
      qa1[2 * j]     = qr[32 + kb];
      qa1[2 * j + 1] = qr[32 + kb + 1];
    }
  }

  float rm[8], rl[8];
#pragma unroll
  for (int g = 0; g < 8; ++g) { rm[g] = -3.0e38f; rl[g] = 0.f; }

  // -------- pass 1: online row max / sum over all 64 key tiles --------
  for (int kt = 0; kt < SEQ / 16; ++kt) {
    v16bf kb0, kb1;
    const __bf16* kr = Kp + (size_t)(kt * 16 + ln) * DHEAD;
#pragma unroll
    for (int j = 0; j < 8; ++j) {
      int kb = kmap(j, half);
      kb0[2 * j]     = kr[kb];
      kb0[2 * j + 1] = kr[kb + 1];
      kb1[2 * j]     = kr[32 + kb];
      kb1[2 * j + 1] = kr[32 + kb + 1];
    }
    if (kt + 1 < SEQ / 16)   // global_prefetch_b8 of next mask rows
      __builtin_prefetch(Mp + (size_t)(q0 + half * 8) * SEQ + (kt + 1) * 16 + ln, 0, 0);
    v8f sc = (v8f){0.f, 0.f, 0.f, 0.f, 0.f, 0.f, 0.f, 0.f};
    sc = wmma_bf16(qa0, kb0, sc);
    sc = wmma_bf16(qa1, kb1, sc);
#pragma unroll
    for (int g = 0; g < 8; ++g) {
      float mv = (Mp[(size_t)(q0 + g + 8 * half) * SEQ + kt * 16 + ln] - 1.0f) * 1.0e9f;
      float s = sc[g] + mv;
      float t = s;
      t = fmaxf(t, __shfl_xor(t, 1));
      t = fmaxf(t, __shfl_xor(t, 2));
      t = fmaxf(t, __shfl_xor(t, 4));
      t = fmaxf(t, __shfl_xor(t, 8));
      float mnew = fmaxf(rm[g], t);
      float e = __expf(s - mnew);
      e += __shfl_xor(e, 1);
      e += __shfl_xor(e, 2);
      e += __shfl_xor(e, 4);
      e += __shfl_xor(e, 8);
      rl[g] = rl[g] * __expf(rm[g] - mnew) + e;
      rm[g] = mnew;
    }
  }

  float rinv[8];
#pragma unroll
  for (int g = 0; g < 8; ++g) rinv[g] = (rl[g] > 0.f) ? 1.0f / rl[g] : 0.f;

  // -------- pass 2: probabilities -> attn store + P @ V (fused) --------
  v8f cacc[4];
#pragma unroll
  for (int dt = 0; dt < 4; ++dt) cacc[dt] = (v8f){0.f, 0.f, 0.f, 0.f, 0.f, 0.f, 0.f, 0.f};

  for (int ktp = 0; ktp < SEQ / 32; ++ktp) {
#pragma unroll
    for (int sub = 0; sub < 2; ++sub) {
      const int kt = ktp * 2 + sub;
      v16bf kb0, kb1;
      const __bf16* kr = Kp + (size_t)(kt * 16 + ln) * DHEAD;
#pragma unroll
      for (int j = 0; j < 8; ++j) {
        int kb = kmap(j, half);
        kb0[2 * j]     = kr[kb];
        kb0[2 * j + 1] = kr[kb + 1];
        kb1[2 * j]     = kr[32 + kb];
        kb1[2 * j + 1] = kr[32 + kb + 1];
      }
      v8f sc = (v8f){0.f, 0.f, 0.f, 0.f, 0.f, 0.f, 0.f, 0.f};
      sc = wmma_bf16(qa0, kb0, sc);
      sc = wmma_bf16(qa1, kb1, sc);
#pragma unroll
      for (int g = 0; g < 8; ++g) {
        float mv = (Mp[(size_t)(q0 + g + 8 * half) * SEQ + kt * 16 + ln] - 1.0f) * 1.0e9f;
        float s = sc[g] + mv;
        float p = __expf(s - rm[g]) * rinv[g];
        Ap[(size_t)(q0 + g + 8 * half) * SEQ + kt * 16 + ln] = p;       // attn out
        Pl[w][g + 8 * half][sub * 16 + ln] = f2bf(p);                   // LDS transpose
      }
    }
    // P fragment in A-layout (DS ops are in-order within a wave)
    v16bf pa;
#pragma unroll
    for (int j = 0; j < 8; ++j) {
      int kb = kmap(j, half);
      pa[2 * j]     = Pl[w][ln][kb];
      pa[2 * j + 1] = Pl[w][ln][kb + 1];
    }
#pragma unroll
    for (int dt = 0; dt < 4; ++dt) {
      v16bf vb;
#pragma unroll
      for (int j = 0; j < 8; ++j) {
        int kb = kmap(j, half);
        int key = ktp * 32 + kb;
        vb[2 * j]     = Vp[(size_t)key * DHEAD + dt * 16 + ln];
        vb[2 * j + 1] = Vp[(size_t)(key + 1) * DHEAD + dt * 16 + ln];
      }
      cacc[dt] = wmma_bf16(pa, vb, cacc[dt]);
    }
  }

  // context -> ws [B,S,1024] bf16 (head-concatenated)
#pragma unroll
  for (int dt = 0; dt < 4; ++dt) {
#pragma unroll
    for (int g = 0; g < 8; ++g) {
      int q = q0 + g + 8 * half;
      Ctx[(size_t)(b * SEQ + q) * F + h * DHEAD + dt * 16 + ln] = f2bf(cacc[dt][g]);
    }
  }
}

// ---------------------------------------------------------------------------
// Kernel 3: out = LN(Ctx @ W_O + X_q).  One block = 16 rows x full 1024 cols.
// grid = 512 blocks, block = 256 (8 waves, each wave owns 128 cols).
// Double-buffered TDM staging of Ctx (bf16) and W_O (fp32, 128KB tiles).
// The LayerNorm row buffer aliases W-buffer 0 (only live after the GEMM).
// Total LDS: 2*128KB + 2*1KB = 258KB (< 320KB/WGP).
// ---------------------------------------------------------------------------
__global__ __launch_bounds__(256) void out_ln_kernel(
    const __bf16* __restrict__ Ctx, const float* __restrict__ Wo,
    const float* __restrict__ Xq, float* __restrict__ Out) {
  __shared__ __align__(16) char smem[2 * (32 * 1024 * 4) + 2 * (16 * 32 * 2)];
  float*  Wl0 = (float*)smem;                               // 128KB
  float*  Wl1 = (float*)(smem + 32 * 1024 * 4);             // 128KB
  __bf16* Al0 = (__bf16*)(smem + 2 * 32 * 1024 * 4);        // 1KB
  __bf16* Al1 = (__bf16*)(smem + 2 * 32 * 1024 * 4 + 16 * 32 * 2);
  float*  Rb  = (float*)smem;   // aliases Wl0 after the GEMM loop (16x1024 f32)

  const int tid = threadIdx.x;
  const int w = tid >> 5, lane = tid & 31;
  const int half = lane >> 4, ln = lane & 15;
  const int m0 = blockIdx.x * 16;

  v8f acc[8];
#pragma unroll
  for (int nt = 0; nt < 8; ++nt) acc[nt] = (v8f){0.f, 0.f, 0.f, 0.f, 0.f, 0.f, 0.f, 0.f};

  if (tid < 32) {   // prologue: stage tile 0
    tdm_load_2d(lds_off(Al0), Ctx + (size_t)m0 * F, 1, 32, 16, F, BATCH * SEQ, F);
    tdm_load_2d(lds_off(Wl0), Wo, 2, 1024, 32, F, F, F);
  }

  for (int it = 0; it < F / 32; ++it) {
    const int buf = it & 1;
    __syncthreads();
    if (tid < 32) {
      if (it + 1 < F / 32) {
        const int k1 = (it + 1) * 32;
        tdm_load_2d(lds_off(buf ? Al0 : Al1), Ctx + (size_t)m0 * F + k1, 1,
                    32, 16, F, BATCH * SEQ, F);
        tdm_load_2d(lds_off(buf ? Wl0 : Wl1), Wo + (size_t)k1 * F, 2,
                    1024, 32, F, F, F);
        __builtin_amdgcn_s_wait_tensorcnt(2);
      } else {
        __builtin_amdgcn_s_wait_tensorcnt(0);
      }
    }
    __syncthreads();

    const __bf16* Af = buf ? Al1 : Al0;
    const float*  Wf = buf ? Wl1 : Wl0;
    v16bf a;
#pragma unroll
    for (int j = 0; j < 8; ++j) {
      int kb = kmap(j, half);
      a[2 * j]     = Af[ln * 32 + kb];
      a[2 * j + 1] = Af[ln * 32 + kb + 1];
    }
#pragma unroll
    for (int nt = 0; nt < 8; ++nt) {
      v16bf bb;
      int n = w * 128 + nt * 16 + ln;
#pragma unroll
      for (int j = 0; j < 8; ++j) {
        int kb = kmap(j, half);
        bb[2 * j]     = f2bf(Wf[kb * 1024 + n]);
        bb[2 * j + 1] = f2bf(Wf[(kb + 1) * 1024 + n]);
      }
      acc[nt] = wmma_bf16(a, bb, acc[nt]);
    }
  }
  __syncthreads();   // GEMM done; Wl0 region now reusable as Rb

  // residual add into LDS row buffer
#pragma unroll
  for (int nt = 0; nt < 8; ++nt) {
#pragma unroll
    for (int g = 0; g < 8; ++g) {
      int r = g + 8 * half;
      int n = w * 128 + nt * 16 + ln;
      Rb[r * 1024 + n] = acc[nt][g] + Xq[(size_t)(m0 + r) * F + n];
    }
  }
  __syncthreads();

  // LayerNorm: 16 threads per row, shuffle-reduce within 16-lane groups
  const int row = tid >> 4, part = tid & 15;
  float s = 0.f, ss = 0.f;
#pragma unroll 8
  for (int j = 0; j < 64; ++j) {
    float v = Rb[row * 1024 + part + j * 16];
    s += v;
    ss += v * v;
  }
  s  += __shfl_xor(s, 1);  s  += __shfl_xor(s, 2);  s  += __shfl_xor(s, 4);  s  += __shfl_xor(s, 8);
  ss += __shfl_xor(ss, 1); ss += __shfl_xor(ss, 2); ss += __shfl_xor(ss, 4); ss += __shfl_xor(ss, 8);
  const float mean = s * (1.0f / 1024.0f);
  const float var  = ss * (1.0f / 1024.0f) - mean * mean;
  const float rstd = rsqrtf(var + 1e-5f);
#pragma unroll 8
  for (int j = 0; j < 64; ++j) {
    int c = part + j * 16;
    float v = Rb[row * 1024 + c];
    Out[(size_t)(m0 + row) * F + c] = (v - mean) * rstd;
  }
}

// ---------------------------------------------------------------------------
extern "C" void kernel_launch(void* const* d_in, const int* in_sizes, int n_in,
                              void* d_out, int out_size, void* d_ws, size_t ws_size,
                              hipStream_t stream) {
  const float* Xq   = (const float*)d_in[0];
  const float* Xk   = (const float*)d_in[1];
  const float* Xv   = (const float*)d_in[2];
  const float* mask = (const float*)d_in[3];
  const float* Wq   = (const float*)d_in[4];
  const float* Wk   = (const float*)d_in[5];
  const float* Wv   = (const float*)d_in[6];
  const float* Wo   = (const float*)d_in[7];

  float* out  = (float*)d_out;                       // [B,S,1024]
  float* attn = out + (size_t)BATCH * SEQ * F;       // [B,H,S,S]

  const size_t QKV_ELEMS = (size_t)BATCH * NH * SEQ * DHEAD;  // 8,388,608
  __bf16* Qb  = (__bf16*)d_ws;
  __bf16* Kb  = Qb + QKV_ELEMS;
  __bf16* Vb  = Kb + QKV_ELEMS;
  __bf16* Ctx = Vb + QKV_ELEMS;                      // [B,S,1024] bf16  (total ws: 64 MB)

  qkv_proj_kernel<<<dim3(F / 64, (BATCH * SEQ) / 128, 3), 256, 0, stream>>>(
      Xq, Xk, Xv, Wq, Wk, Wv, Qb, Kb, Vb);
  attn_kernel<<<dim3((BATCH * NH * (SEQ / 16)) / 8), 256, 0, stream>>>(
      Qb, Kb, Vb, mask, attn, Ctx);
  out_ln_kernel<<<dim3((BATCH * SEQ) / 16), 256, 0, stream>>>(Ctx, Wo, Xq, out);
}
